// TransformerLayer_54451595378769
// MI455X (gfx1250) — matrix operand order repaired and checked by reading, not compiled, not run on hardware
//
#include <hip/hip_runtime.h>
#include <hip/hip_bf16.h>

typedef __attribute__((ext_vector_type(16))) _Float16 v16h;
typedef __attribute__((ext_vector_type(8)))  _Float16 v8h;
typedef __attribute__((ext_vector_type(8)))  float    v8f;

#define DIM  1024
#define SEQ  2048
#define NTOK 4096   // B * S
#define NH   16
#define HD   64
#define MLP  4096

static __device__ __forceinline__ v16h cat8(v8h lo, v8h hi) {
  v16h r;
#pragma unroll
  for (int i = 0; i < 8; ++i) { r[i] = lo[i]; r[i + 8] = hi[i]; }
  return r;
}

static __device__ __forceinline__ v8f wmma16(v16h a, v16h b, v8f c) {
  return __builtin_amdgcn_wmma_f32_16x16x32_f16(false, a, false, b,
                                                (short)0, c, false, false);
}

// low 32 bits of a generic LDS pointer == wave-relative LDS byte offset
static __device__ __forceinline__ unsigned ldsoff(const void* p) {
  return (unsigned)(uintptr_t)p;
}

// async DMA: 16 bytes global -> LDS, tracked by ASYNCcnt (CDNA5)
static __device__ __forceinline__ void async_b128(const void* g, unsigned lds) {
  asm volatile("global_load_async_to_lds_b128 %0, %1, off"
               :: "v"(lds), "v"((unsigned long long)(uintptr_t)g)
               : "memory");
}
#define WAIT_ASYNC(n) asm volatile("s_wait_asynccnt %0" :: "n"(n) : "memory")

// DPP16 row_xmask lane swap (VALU, no LDS): lane -> lane ^ MASK within row-of-16
template <int CTRL>
static __device__ __forceinline__ float dpp_movf(float v) {
  int i = __float_as_int(v);
  int r = __builtin_amdgcn_update_dpp(i, i, CTRL, 0xf, 0xf, true);
  return __int_as_float(r);
}
// max across the 16 lanes of a lane-half (rows stay within their half)
static __device__ __forceinline__ float rowmax16(float v) {
  v = fmaxf(v, dpp_movf<0x161>(v));  // row_xmask:1
  v = fmaxf(v, dpp_movf<0x162>(v));  // row_xmask:2
  v = fmaxf(v, dpp_movf<0x164>(v));  // row_xmask:4
  v = fmaxf(v, dpp_movf<0x168>(v));  // row_xmask:8
  return v;
}

// ---------------- weight convert (f32 [K,N] -> f16 [N,K]) ----------------
__global__ void wcvt_kernel(const float* __restrict__ w,
                            _Float16* __restrict__ wt, int K, int N) {
  int idx = blockIdx.x * 256 + threadIdx.x;
  if (idx >= K * N) return;
  int k = idx / N, n = idx % N;
  wt[(size_t)n * K + k] = (_Float16)w[idx];
}

// ---------------- layernorm, f32 in -> f16 out ----------------
__global__ __launch_bounds__(256) void ln_kernel(const float* __restrict__ x,
                                                 const float* __restrict__ g,
                                                 const float* __restrict__ b,
                                                 _Float16* __restrict__ out) {
  __shared__ float s1[256], s2[256];
  const int row = blockIdx.x;
  const float* xr = x + (size_t)row * DIM;
  float sum = 0.f, sq = 0.f;
#pragma unroll
  for (int i = 0; i < 4; ++i) {
    float v = xr[threadIdx.x + i * 256];
    sum += v; sq += v * v;
  }
  s1[threadIdx.x] = sum; s2[threadIdx.x] = sq;
  __syncthreads();
  for (int off = 128; off > 0; off >>= 1) {
    if (threadIdx.x < off) {
      s1[threadIdx.x] += s1[threadIdx.x + off];
      s2[threadIdx.x] += s2[threadIdx.x + off];
    }
    __syncthreads();
  }
  const float mu  = s1[0] * (1.0f / DIM);
  const float var = s2[0] * (1.0f / DIM) - mu * mu;
  const float rs  = rsqrtf(var + 1e-5f);
#pragma unroll
  for (int i = 0; i < 4; ++i) {
    int c = threadIdx.x + i * 256;
    out[(size_t)row * DIM + c] = (_Float16)((xr[c] - mu) * rs * g[c] + b[c]);
  }
}

// ------- tiled WMMA GEMM, async double-buffered: C = A[M,K] * Bt[N,K]^T -------
// 256 threads = 8 waves; block tile 128x128, BK=64; wave tile 64x32.
// 16 WMMAs per wave between barriers; LDS 2 x (16+16) KB = 64 KB.
template <bool RELU, bool RESID, bool OUTF16>
__global__ __launch_bounds__(256) void gemm_kernel(
    const _Float16* __restrict__ A, const _Float16* __restrict__ Bt,
    const float* __restrict__ bias, const float* __restrict__ res,
    void* __restrict__ outp, int M, int N, int K) {
  __shared__ alignas(32) _Float16 As[2][128][64];
  __shared__ alignas(32) _Float16 Bs[2][128][64];
  const int tid  = threadIdx.x;
  const int lane = tid & 31;
  const int wave = tid >> 5;
  const int wm   = wave & 1;   // M half (64)
  const int wn   = wave >> 1;  // N quarter (32)
  const int m0   = blockIdx.y * 128;
  const int n0   = blockIdx.x * 128;

  const int srow = tid >> 1;          // staging row 0..127 (for both A and Bt)
  const int skc  = (tid & 1) * 32;    // k-chunk (32 f16 = 64B, four b128 asyncs)

  const _Float16* ag = &A[(size_t)(m0 + srow) * K + skc];
  const _Float16* bg = &Bt[(size_t)(n0 + srow) * K + skc];

  // prologue: stage k-step 0 into buffer 0
#pragma unroll
  for (int q = 0; q < 4; ++q) {
    async_b128(ag + q * 8, ldsoff(&As[0][srow][skc + q * 8]));
    async_b128(bg + q * 8, ldsoff(&Bs[0][srow][skc + q * 8]));
  }
  WAIT_ASYNC(0);
  __syncthreads();

  v8f acc[4][2] = {};
  const int aoff = (lane < 16) ? 0 : 16;   // A-operand contiguous halves
  const int boff = (lane < 16) ? 0 : 16;   // B-operand contiguous halves
  const int nk = K / 64;

  for (int i = 0; i < nk; ++i) {
    const int cur = i & 1;
    if (i + 1 < nk) {  // async-prefetch next k-step into the other buffer
      const _Float16* an = ag + (size_t)(i + 1) * 64;
      const _Float16* bn = bg + (size_t)(i + 1) * 64;
#pragma unroll
      for (int q = 0; q < 4; ++q) {
        async_b128(an + q * 8, ldsoff(&As[cur ^ 1][srow][skc + q * 8]));
        async_b128(bn + q * 8, ldsoff(&Bs[cur ^ 1][srow][skc + q * 8]));
      }
    }

#pragma unroll
    for (int kk = 0; kk < 2; ++kk) {  // two K=32 sub-steps
      v16h afrag[4], bfrag[2];
#pragma unroll
      for (int ii = 0; ii < 4; ++ii) {
        int r = wm * 64 + ii * 16 + (lane & 15);
        afrag[ii] = *(const v16h*)&As[cur][r][kk * 32 + aoff];
      }
#pragma unroll
      for (int j = 0; j < 2; ++j) {
        int c = wn * 32 + j * 16 + (lane & 15);
        bfrag[j] = *(const v16h*)&Bs[cur][c][kk * 32 + boff];
      }
#pragma unroll
      for (int ii = 0; ii < 4; ++ii)
#pragma unroll
        for (int j = 0; j < 2; ++j)
          acc[ii][j] = wmma16(afrag[ii], bfrag[j], acc[ii][j]);
    }

    WAIT_ASYNC(0);     // next buffer's DMA landed (this wave's share)
    __syncthreads();   // all waves done reading cur + all DMA visible
  }

#pragma unroll
  for (int ii = 0; ii < 4; ++ii)
#pragma unroll
    for (int j = 0; j < 2; ++j)
#pragma unroll
      for (int r = 0; r < 8; ++r) {
        int row = m0 + wm * 64 + ii * 16 + r + ((lane < 16) ? 0 : 8);
        int col = n0 + wn * 32 + j * 16 + (lane & 15);
        float v = acc[ii][j][r] + bias[col];
        if (RELU)  v = fmaxf(v, 0.0f);
        if (RESID) v += res[(size_t)row * N + col];
        if (OUTF16) ((_Float16*)outp)[(size_t)row * N + col] = (_Float16)v;
        else        ((float*)outp)[(size_t)row * N + col] = v;
      }
}

// ---------------- flash attention (online softmax, WMMA f16) ----------------
// grid: (SEQ/64, NH, B); 128 threads = 4 waves, each wave owns 16 query rows.
// K staged via async DMA, V via registers (transposed, packed b32 stores),
// double-buffered, one barrier per 32-key step. Row-max via DPP row_xmask,
// row-sum via an extra WMMA against a ones matrix.
__global__ __launch_bounds__(128) void attn_kernel(const _Float16* __restrict__ qkv,
                                                   _Float16* __restrict__ ctx) {
  __shared__ alignas(32) _Float16 Ks[2][32][64];       // [key][dim]
  __shared__ alignas(32) _Float16 Vt[2][64][32];       // [dim][key] transposed
  __shared__ alignas(32) _Float16 Psh[4][16 * 32];     // per-wave P transpose
  const int tid  = threadIdx.x;
  const int lane = tid & 31;
  const int wv   = tid >> 5;
  const int qt   = blockIdx.x;
  const int h    = blockIdx.y;
  const int bb   = blockIdx.z;
  const size_t tokbase = (size_t)bb * SEQ;
  const int qoff = h * HD;
  const int koff = DIM + h * HD;
  const int voff = 2 * DIM + h * HD;
  const int aoff = (lane < 16) ? 0 : 16;

  // staging maps
  const int sk_key = tid >> 2, sk_d = (tid & 3) * 16;      // K: 32B per thread
  const int vp_key = (tid >> 3) * 2, vp_d = (tid & 7) * 8; // V: key pair x 8 dims

  // ones matrix for WMMA row-sum reduction
  v16h ones;
#pragma unroll
  for (int e = 0; e < 16; ++e) ones[e] = (_Float16)1.0f;

  // Q fragment (16 rows x 64 dims, two K=32 chunks), loaded once
  const int qrow = qt * 64 + wv * 16 + (lane & 15);
  const _Float16* qp = qkv + (tokbase + qrow) * (size_t)(3 * DIM) + qoff;
  v16h aq[2];
#pragma unroll
  for (int c = 0; c < 2; ++c)
    aq[c] = *(const v16h*)&qp[c * 32 + aoff];

  // prologue: stage key-block 0 into buffer 0
  {
    const _Float16* kg = &qkv[(tokbase + sk_key) * (size_t)(3 * DIM) + koff + sk_d];
    async_b128(kg,     ldsoff(&Ks[0][sk_key][sk_d]));
    async_b128(kg + 8, ldsoff(&Ks[0][sk_key][sk_d + 8]));
    v8h va = *(const v8h*)&qkv[(tokbase + vp_key) * (size_t)(3 * DIM) + voff + vp_d];
    v8h vb = *(const v8h*)&qkv[(tokbase + vp_key + 1) * (size_t)(3 * DIM) + voff + vp_d];
#pragma unroll
    for (int e = 0; e < 8; ++e) {
      union { _Float16 hh[2]; unsigned u; } pk;
      pk.hh[0] = va[e]; pk.hh[1] = vb[e];
      *(unsigned*)&Vt[0][vp_d + e][vp_key] = pk.u;
    }
    WAIT_ASYNC(0);
    __syncthreads();
  }

  v8f o[4] = {};
  float mrun[8], lrun[8];
#pragma unroll
  for (int r = 0; r < 8; ++r) { mrun[r] = -1e30f; lrun[r] = 0.0f; }

  const int nblk = SEQ / 32;
  for (int i = 0; i < nblk; ++i) {
    const int cur = i & 1;
    const bool hasnext = (i + 1 < nblk);
    v8h va, vb;
    if (hasnext) {  // prefetch: K via async DMA, V into registers
      const size_t jn = tokbase + (size_t)(i + 1) * 32;
      const _Float16* kg = &qkv[(jn + sk_key) * (size_t)(3 * DIM) + koff + sk_d];
      async_b128(kg,     ldsoff(&Ks[cur ^ 1][sk_key][sk_d]));
      async_b128(kg + 8, ldsoff(&Ks[cur ^ 1][sk_key][sk_d + 8]));
      va = *(const v8h*)&qkv[(jn + vp_key) * (size_t)(3 * DIM) + voff + vp_d];
      vb = *(const v8h*)&qkv[(jn + vp_key + 1) * (size_t)(3 * DIM) + voff + vp_d];
    }

    // scores: 16 q-rows x 32 keys as two 16x16 tiles
    v8f s0 = {}, s1 = {};
#pragma unroll
    for (int c = 0; c < 2; ++c) {
      int keyl = lane & 15;
      int kd   = c * 32 + ((lane < 16) ? 0 : 16);
      v16h bk0 = *(const v16h*)&Ks[cur][keyl][kd];
      v16h bk1 = *(const v16h*)&Ks[cur][16 + keyl][kd];
      s0 = wmma16(aq[c], bk0, s0);
      s1 = wmma16(aq[c], bk1, s1);
    }

    // online softmax: row max via DPP butterflies (no LDS traffic)
    const float scale = 0.125f;  // 1/sqrt(64)
    float alpha[8];
    v8f p0, p1;
#pragma unroll
    for (int r = 0; r < 8; ++r) {
      float a = s0[r] * scale, bvv = s1[r] * scale;
      float m = rowmax16(fmaxf(a, bvv));
      float mnew = fmaxf(mrun[r], m);
      alpha[r] = __expf(mrun[r] - mnew);
      p0[r] = __expf(a - mnew);
      p1[r] = __expf(bvv - mnew);
      mrun[r] = mnew;
    }
#pragma unroll
    for (int t = 0; t < 4; ++t)
#pragma unroll
      for (int r = 0; r < 8; ++r) o[t][r] *= alpha[r];

    // P: C-layout -> A-operand layout via per-wave LDS (same-wave: no barrier,
    // the backend inserts s_wait_dscnt for the RAW dependence)
#pragma unroll
    for (int r = 0; r < 8; ++r) {
      int row = r + ((lane < 16) ? 0 : 8);
      int col = lane & 15;
      Psh[wv][row * 32 + col]      = (_Float16)p0[r];
      Psh[wv][row * 32 + 16 + col] = (_Float16)p1[r];
    }
    int m16 = lane & 15;
    v16h pf = *(const v16h*)&Psh[wv][m16 * 32 + aoff];

    // row-sum of P via matrix unit: P(16x32) @ ones(32x16) -> every col = rowsum
    v8f psum = wmma16(pf, ones, v8f{});
#pragma unroll
    for (int r = 0; r < 8; ++r) lrun[r] = lrun[r] * alpha[r] + psum[r];

    // O += P(16x32) @ V(32x64) : 4 N-tiles
    int ko = (lane < 16) ? 0 : 16;
#pragma unroll
    for (int t = 0; t < 4; ++t) {
      v16h bv = *(const v16h*)&Vt[cur][t * 16 + (lane & 15)][ko];
      o[t] = wmma16(pf, bv, o[t]);
    }

    if (hasnext) {  // transpose-store prefetched V into the other buffer
#pragma unroll
      for (int e = 0; e < 8; ++e) {
        union { _Float16 hh[2]; unsigned u; } pk;
        pk.hh[0] = va[e]; pk.hh[1] = vb[e];
        *(unsigned*)&Vt[cur ^ 1][vp_d + e][vp_key] = pk.u;
      }
    }
    WAIT_ASYNC(0);
    __syncthreads();  // single barrier per key-block
  }

  // normalize + write ctx [B,S,D] f16
#pragma unroll
  for (int t = 0; t < 4; ++t)
#pragma unroll
    for (int r = 0; r < 8; ++r) {
      int row = qt * 64 + wv * 16 + r + ((lane < 16) ? 0 : 8);
      int col = h * HD + t * 16 + (lane & 15);
      ctx[(tokbase + row) * (size_t)DIM + col] = (_Float16)(o[t][r] / lrun[r]);
    }
}

// ---------------------------------------------------------------------------
extern "C" void kernel_launch(void* const* d_in, const int* in_sizes, int n_in,
                              void* d_out, int out_size, void* d_ws, size_t ws_size,
                              hipStream_t stream) {
  (void)in_sizes; (void)n_in; (void)out_size; (void)ws_size;
  const float* x     = (const float*)d_in[0];
  const float* w_qkv = (const float*)d_in[1];
  const float* b_qkv = (const float*)d_in[2];
  const float* w_out = (const float*)d_in[3];
  const float* b_out = (const float*)d_in[4];
  const float* w_fc1 = (const float*)d_in[5];
  const float* b_fc1 = (const float*)d_in[6];
  const float* w_fc2 = (const float*)d_in[7];
  const float* b_fc2 = (const float*)d_in[8];
  const float* ln1_g = (const float*)d_in[9];
  const float* ln1_b = (const float*)d_in[10];
  const float* ln2_g = (const float*)d_in[11];
  const float* ln2_b = (const float*)d_in[12];

  char* ws = (char*)d_ws;
  size_t off = 0;
  auto alloc = [&](size_t bytes) {
    void* p = ws + off;
    off += (bytes + 255) & ~(size_t)255;
    return p;
  };
  _Float16* wq16 = (_Float16*)alloc((size_t)DIM * 3 * DIM * 2);
  _Float16* wo16 = (_Float16*)alloc((size_t)DIM * DIM * 2);
  _Float16* w116 = (_Float16*)alloc((size_t)DIM * MLP * 2);
  _Float16* w216 = (_Float16*)alloc((size_t)MLP * DIM * 2);
  _Float16* h16  = (_Float16*)alloc((size_t)NTOK * DIM * 2);
  // qkv16 (live through attention) and ffn16 (live after) share one region
  _Float16* qkv16 = (_Float16*)alloc((size_t)NTOK * MLP * 2);  // >= NTOK*3*DIM
  _Float16* ffn16 = qkv16;
  _Float16* ctx16 = (_Float16*)alloc((size_t)NTOK * DIM * 2);
  float*    x1    = (float*)alloc((size_t)NTOK * DIM * 4);

  // 1. weight f32->f16 transpose
  wcvt_kernel<<<(DIM * 3 * DIM) / 256, 256, 0, stream>>>(w_qkv, wq16, DIM, 3 * DIM);
  wcvt_kernel<<<(DIM * DIM) / 256, 256, 0, stream>>>(w_out, wo16, DIM, DIM);
  wcvt_kernel<<<(DIM * MLP) / 256, 256, 0, stream>>>(w_fc1, w116, DIM, MLP);
  wcvt_kernel<<<(MLP * DIM) / 256, 256, 0, stream>>>(w_fc2, w216, MLP, DIM);

  // 2. LN1 -> h16
  ln_kernel<<<NTOK, 256, 0, stream>>>(x, ln1_g, ln1_b, h16);
  // 3. qkv = h @ w_qkv + b_qkv  (f16 out)
  gemm_kernel<false, false, true><<<dim3(3 * DIM / 128, NTOK / 128), 256, 0, stream>>>(
      h16, wq16, b_qkv, nullptr, qkv16, NTOK, 3 * DIM, DIM);
  // 4. flash attention -> ctx16
  attn_kernel<<<dim3(SEQ / 64, NH, 2), 128, 0, stream>>>(qkv16, ctx16);
  // 5. x1 = x + ctx @ w_out + b_out  (f32 out, residual)
  gemm_kernel<false, true, false><<<dim3(DIM / 128, NTOK / 128), 256, 0, stream>>>(
      ctx16, wo16, b_out, x, x1, NTOK, DIM, DIM);
  // 6. LN2 -> h16
  ln_kernel<<<NTOK, 256, 0, stream>>>(x1, ln2_g, ln2_b, h16);
  // 7. ffn = relu(h @ w_fc1 + b_fc1)  (f16 out)
  gemm_kernel<true, false, true><<<dim3(MLP / 128, NTOK / 128), 256, 0, stream>>>(
      h16, w116, b_fc1, nullptr, ffn16, NTOK, MLP, DIM);
  // 8. out = x1 + ffn @ w_fc2 + b_fc2  (f32 out, residual)
  gemm_kernel<false, true, false><<<dim3(DIM / 128, NTOK / 128), 256, 0, stream>>>(
      ffn16, w216, b_fc2, x1, (float*)d_out, NTOK, DIM, MLP);
}
